// CCAttention_11115375362197
// MI455X (gfx1250) — matrix operand order repaired and checked by reading, not compiled, not run on hardware
//
#include <hip/hip_runtime.h>

// ---------------------------------------------------------------------------
// CDNA5 (gfx1250) criss-cross attention block.
// Heavy math: v_wmma_f32_16x16x32_bf16 (fp32 accumulate).
// A-tiles staged global->LDS with GLOBAL_LOAD_ASYNC_TO_LDS_B128 (ASYNCcnt),
// double-buffered: s_wait_asynccnt 2 overlaps tile i+1 DMA with tile i WMMAs.
// B-tiles register-staged one iteration ahead; parked in LDS as ds_store_b32.
// All 8 B fragments pre-loaded so the 8 WMMAs issue back-to-back.
// ---------------------------------------------------------------------------

typedef __attribute__((ext_vector_type(16))) __bf16 v16bf;
typedef __attribute__((ext_vector_type(8)))  __bf16 v8bf;
typedef __attribute__((ext_vector_type(2)))  __bf16 v2bf;
typedef __attribute__((ext_vector_type(8)))  float  v8f;

static constexpr int  BATCH = 2;
static constexpr int  CIN   = 2048;
static constexpr int  C     = 512;    // CINT
static constexpr int  CQK   = 64;
static constexpr int  H     = 96;
static constexpr int  W     = 96;
static constexpr long HWX   = (long)H * W;        // 9216
static constexpr int  NG    = 32;

// GEMM flags
static constexpr int FLAG_MASK_DIAG = 1;  // C[m][n] = -1e30 when m==n (e_h diag)
static constexpr int FLAG_ACCUM     = 2;  // C += result
static constexpr int FLAG_ASYNC_A   = 4;  // Ak==1: stage A tile via async->LDS
static constexpr int FLAG_BN1       = 8;  // Bn==1 && N%128==0: vector B loads

struct GemmP {
  const __bf16* A;      // element (m,k) at A + m*Am + k*Ak (+ slice offsets)
  const __bf16* B;      // element (k,n) at B + k*Bk + n*Bn
  const float*  bias;   // optional, indexed by m
  float*        Cf;     // optional f32 output
  __bf16*       Cb;     // optional bf16 output
  long Am, Ak, Ain, Aout;
  long Bk, Bn, Bin, Bout;
  long Cm, Cn, Cin, Cout;
  int M, N, K;
  int D0;               // slice s -> s0 = s % D0 (inner), s1 = s / D0 (outer)
  int slices;
  int flags;
};

__device__ __forceinline__ void async_a_b128(const __bf16* gaddr, unsigned lds_off) {
  // GLOBAL_LOAD_ASYNC_TO_LDS_B128: per-lane 16B global -> per-lane LDS addr.
  asm volatile("global_load_async_to_lds_b128 %0, %1, off"
               :: "v"(lds_off), "v"(gaddr) : "memory");
}
__device__ __forceinline__ void wait_async(int n) {
  if (n == 0) asm volatile("s_wait_asynccnt 0x0" ::: "memory");
  else        asm volatile("s_wait_asynccnt 0x2" ::: "memory");
}

// ---------------------------------------------------------------------------
// Generic strided bf16 WMMA GEMM.
// Block: 256 threads = 8 waves. Block tile: M=128 (16 rows/wave), N=128.
// Per K-step (32): 8 v_wmma per wave (one A fragment reused over 8 N-chunks).
// Double-buffered LDS; async-A DMA (Ak==1) overlapped with compute.
// ---------------------------------------------------------------------------
__global__ __launch_bounds__(256) void wmma_gemm(GemmP p) {
  __shared__ __attribute__((aligned(16))) __bf16 sA[2][128 * 32];  // [row][k]
  __shared__ __attribute__((aligned(16))) __bf16 sB[2][128 * 32];  // [col][k]

  const int tid  = threadIdx.x;
  const int wave = tid >> 5;
  const int lane = tid & 31;
  const int tileN = blockIdx.x * 128;
  const int tileM = blockIdx.y * 128;
  const int s  = blockIdx.z;
  const int s0 = s % p.D0;
  const int s1 = s / p.D0;
  const bool asyncA = (p.flags & FLAG_ASYNC_A) != 0;

  const __bf16* A  = p.A + (long)s1 * p.Aout + (long)s0 * p.Ain;
  const __bf16* Bp = p.B + (long)s1 * p.Bout + (long)s0 * p.Bin;

  v8f acc[8] = {};

  // A fragment (16-bit A 16x32, ISA 7.12.2): lanes 0-15 row=lane, K {0..7,16..23};
  // lanes 16-31 row=lane-16, K {8..15,24..31}.
  int rowA = tileM + wave * 16 + (lane & 15);
  if (rowA >= p.M) rowA = p.M - 1;
  const int rowLoc  = wave * 16 + (lane & 15);
  const int kbaseA  = (lane < 16) ? 0 : 8;
  // B fragment (16-bit B 32x16): lanes 0-15 col=lane K {0..7,16..23}; 16-31 K {8..15,24..31}.
  const int colB    = lane & 15;
  const int kbaseB  = (lane < 16) ? 0 : 8;

  // This thread's B-tile slice: 2 k-rows x 8 cols -> 8 aligned b32 LDS stores.
  const int kkB = (tid >> 4) * 2;     // even, 0..30
  const int ccB = (tid & 15) * 8;     // multiple of 8, 0..120
  // This thread's A-tile async slices (2 x B128 = 2 x 8 halves).
  const int rA0 = (tid * 2) >> 2, sA0 = ((tid * 2) & 3) * 8;
  const int rA1 = (tid * 2 + 1) >> 2, sA1 = ((tid * 2 + 1) & 3) * 8;
  int gr0 = tileM + rA0; if (gr0 >= p.M) gr0 = p.M - 1;
  int gr1 = tileM + rA1; if (gr1 >= p.M) gr1 = p.M - 1;

  auto issueA = [&](int k0, int buf) {
    async_a_b128(A + (long)gr0 * p.Am + (k0 + sA0),
                 (unsigned)(size_t)&sA[buf][rA0 * 32 + sA0]);
    async_a_b128(A + (long)gr1 * p.Am + (k0 + sA1),
                 (unsigned)(size_t)&sA[buf][rA1 * 32 + sA1]);
  };
  // breg[0..7] = row kkB cols ccB..+7 ; breg[8..15] = row kkB+1.
  auto loadB = [&](int k0, v16bf& out) {
    if (p.flags & FLAG_BN1) {
      const v8bf r0 = *(const v8bf*)(Bp + (long)(k0 + kkB)     * p.Bk + (tileN + ccB));
      const v8bf r1 = *(const v8bf*)(Bp + (long)(k0 + kkB + 1) * p.Bk + (tileN + ccB));
      #pragma unroll
      for (int j = 0; j < 8; ++j) { out[j] = r0[j]; out[8 + j] = r1[j]; }
    } else {
      #pragma unroll
      for (int j = 0; j < 8; ++j) {
        int col = tileN + ccB + j; if (col >= p.N) col = p.N - 1;
        out[j]     = Bp[(long)(k0 + kkB)     * p.Bk + (long)col * p.Bn];
        out[8 + j] = Bp[(long)(k0 + kkB + 1) * p.Bk + (long)col * p.Bn];
      }
    }
  };

  const int steps = p.K / 32;
  v16bf breg;
  if (asyncA) issueA(0, 0);
  loadB(0, breg);

  for (int i = 0; i < steps; ++i) {
    const int pb = i & 1;
    const int k0 = i * 32;

    // Park tile i's B data in LDS (b32 pairs); start tile i+1's DMA + loads.
    #pragma unroll
    for (int j = 0; j < 8; ++j) {
      v2bf pr; pr[0] = breg[j]; pr[1] = breg[8 + j];
      *(v2bf*)&sB[pb][(ccB + j) * 32 + kkB] = pr;
    }
    const bool more = (i + 1) < steps;
    if (more) {
      if (asyncA) issueA(k0 + 32, 1 - pb);
      loadB(k0 + 32, breg);
    }
    if (asyncA) wait_async(more ? 2 : 0);  // tile i's async loads complete
    __syncthreads();                       // tile i visible to all waves

    v16bf af;
    if (asyncA) {
      const __bf16* ap = &sA[pb][rowLoc * 32 + kbaseA];
      #pragma unroll
      for (int j = 0; j < 8; ++j) { af[j] = ap[j]; af[j + 8] = ap[16 + j]; }
    } else {
      #pragma unroll
      for (int j = 0; j < 8; ++j) {
        af[j]     = A[(long)rowA * p.Am + (long)(k0 + kbaseA + j)      * p.Ak];
        af[j + 8] = A[(long)rowA * p.Am + (long)(k0 + kbaseA + 16 + j) * p.Ak];
      }
    }

    // Pre-load all 8 B fragments (independent regs), then WMMAs back-to-back.
    v16bf bfr[8];
    #pragma unroll
    for (int c = 0; c < 8; ++c) {
      #pragma unroll
      for (int j = 0; j < 16; ++j)
        bfr[c][j] = sB[pb][(c * 16 + colB) * 32 + kbaseB + (j & 7) + ((j >> 3) << 4)];
    }
    #pragma unroll
    for (int c = 0; c < 8; ++c)
      acc[c] = __builtin_amdgcn_wmma_f32_16x16x32_bf16(
          false, af, false, bfr[c], (short)0, acc[c], false, false);
    __syncthreads();   // all waves done with buffer pb before it is re-staged
  }

  // Epilogue. C layout: VGPR i -> M = i (+8 for lanes 16-31); N = lane&15.
  float*  Cf = p.Cf ? p.Cf + (long)s1 * p.Cout + (long)s0 * p.Cin : nullptr;
  __bf16* Cb = p.Cb ? p.Cb + (long)s1 * p.Cout + (long)s0 * p.Cin : nullptr;
  const int rbase = tileM + wave * 16 + ((lane < 16) ? 0 : 8);
  #pragma unroll
  for (int c = 0; c < 8; ++c) {
    const int col = tileN + c * 16 + (lane & 15);
    if (col >= p.N) continue;
    #pragma unroll
    for (int i = 0; i < 8; ++i) {
      const int row = rbase + i;
      if (row >= p.M) continue;
      float v = acc[c][i];
      if (p.bias) v += p.bias[row];
      if ((p.flags & FLAG_MASK_DIAG) && row == col) v = -1e30f;
      const long off = (long)row * p.Cm + (long)col * p.Cn;
      if (p.flags & FLAG_ACCUM) v += Cf[off];
      if (Cf) Cf[off] = v;
      if (Cb) Cb[off] = (__bf16)v;
    }
  }
}

// ---------------------------------------------------------------------------
// Implicit-GEMM 3x3 conv: M=512 (o), N=9216 (h*w), K=4608 (c*9 + tap).
// Weights K-contiguous -> async-to-LDS staging (double-buffered); borders
// zero-padded in the register-staged B loader.
// ---------------------------------------------------------------------------
__global__ __launch_bounds__(256) void wmma_conv3x3(const __bf16* __restrict__ in,
                                                    const __bf16* __restrict__ wt,
                                                    float* __restrict__ out) {
  __shared__ __attribute__((aligned(16))) __bf16 sA[2][128 * 32];
  __shared__ __attribute__((aligned(16))) __bf16 sB[2][128 * 32];

  const int tid  = threadIdx.x;
  const int wave = tid >> 5;
  const int lane = tid & 31;
  const int tileN = blockIdx.x * 128;    // N = 9216 = 72 * 128
  const int tileM = blockIdx.y * 128;    // M = 512
  const __bf16* inb = in  + (long)blockIdx.z * C * HWX;
  float*       outb = out + (long)blockIdx.z * C * HWX;

  v8f acc[8] = {};
  const int rowLoc = wave * 16 + (lane & 15);
  const int kbaseA = (lane < 16) ? 0 : 8;
  const int colB   = lane & 15;
  const int kbaseB = (lane < 16) ? 0 : 8;

  const int kkB = (tid >> 4) * 2;
  const int ccB = (tid & 15) * 8;
  const int rA0 = (tid * 2) >> 2, sA0 = ((tid * 2) & 3) * 8;
  const int rA1 = (tid * 2 + 1) >> 2, sA1 = ((tid * 2 + 1) & 3) * 8;

  auto issueA = [&](int k0, int buf) {
    async_a_b128(wt + (long)(tileM + rA0) * (C * 9) + (k0 + sA0),
                 (unsigned)(size_t)&sA[buf][rA0 * 32 + sA0]);
    async_a_b128(wt + (long)(tileM + rA1) * (C * 9) + (k0 + sA1),
                 (unsigned)(size_t)&sA[buf][rA1 * 32 + sA1]);
  };
  auto loadBrow = [&](int kg, v16bf& o, int base) {
    const int ch  = kg / 9;
    const int tap = kg % 9;
    const int dy  = tap / 3 - 1, dx = tap % 3 - 1;
    #pragma unroll
    for (int j = 0; j < 8; ++j) {
      const int n  = tileN + ccB + j;
      const int hh = n / W + dy, ww = n % W + dx;
      __bf16 v = (__bf16)0.0f;
      if (hh >= 0 && hh < H && ww >= 0 && ww < W)
        v = inb[(long)ch * HWX + hh * W + ww];
      o[base + j] = v;
    }
  };

  const int steps = (C * 9) / 32;
  v16bf breg;
  issueA(0, 0);
  loadBrow(kkB, breg, 0);
  loadBrow(kkB + 1, breg, 8);

  for (int i = 0; i < steps; ++i) {
    const int pb = i & 1;
    #pragma unroll
    for (int j = 0; j < 8; ++j) {
      v2bf pr; pr[0] = breg[j]; pr[1] = breg[8 + j];
      *(v2bf*)&sB[pb][(ccB + j) * 32 + kkB] = pr;
    }
    const bool more = (i + 1) < steps;
    if (more) {
      issueA(i * 32 + 32, 1 - pb);
      loadBrow(i * 32 + 32 + kkB, breg, 0);
      loadBrow(i * 32 + 32 + kkB + 1, breg, 8);
    }
    wait_async(more ? 2 : 0);
    __syncthreads();

    v16bf af;
    const __bf16* ap = &sA[pb][rowLoc * 32 + kbaseA];
    #pragma unroll
    for (int j = 0; j < 8; ++j) { af[j] = ap[j]; af[j + 8] = ap[16 + j]; }

    v16bf bfr[8];
    #pragma unroll
    for (int c = 0; c < 8; ++c) {
      #pragma unroll
      for (int j = 0; j < 16; ++j)
        bfr[c][j] = sB[pb][(c * 16 + colB) * 32 + kbaseB + (j & 7) + ((j >> 3) << 4)];
    }
    #pragma unroll
    for (int c = 0; c < 8; ++c)
      acc[c] = __builtin_amdgcn_wmma_f32_16x16x32_bf16(
          false, af, false, bfr[c], (short)0, acc[c], false, false);
    __syncthreads();
  }

  const int rbase = tileM + wave * 16 + ((lane < 16) ? 0 : 8);
  #pragma unroll
  for (int c = 0; c < 8; ++c) {
    const int col = tileN + c * 16 + (lane & 15);
    #pragma unroll
    for (int i = 0; i < 8; ++i)
      outb[(long)(rbase + i) * HWX + col] = acc[c][i];
  }
}

// ---------------------------------------------------------------------------
// Softmax over 192 logits per pixel (wave-per-pixel, wave32 shuffles).
// ---------------------------------------------------------------------------
__global__ __launch_bounds__(256) void softmax192(const float* __restrict__ E,
                                                  __bf16* __restrict__ P,
                                                  int npix) {
  const int gw   = (int)((blockIdx.x * (long)blockDim.x + threadIdx.x) >> 5);
  const int lane = threadIdx.x & 31;
  if (gw >= npix) return;
  const float* e = E + (long)gw * 192;
  __bf16*      p = P + (long)gw * 192;
  float v[6], m = -3.0e38f;
  #pragma unroll
  for (int i = 0; i < 6; ++i) { v[i] = e[lane + i * 32]; m = fmaxf(m, v[i]); }
  #pragma unroll
  for (int off = 16; off; off >>= 1) m = fmaxf(m, __shfl_xor(m, off, 32));
  float s = 0.0f;
  #pragma unroll
  for (int i = 0; i < 6; ++i) { v[i] = __expf(v[i] - m); s += v[i]; }
  #pragma unroll
  for (int off = 16; off; off >>= 1) s += __shfl_xor(s, off, 32);
  const float inv = 1.0f / s;
  #pragma unroll
  for (int i = 0; i < 6; ++i) p[lane + i * 32] = (__bf16)(v[i] * inv);
}

// cur = gamma*attn + cur ; refresh bf16 mirror
__global__ void combine_residual(const float* __restrict__ attn,
                                 const float* __restrict__ gamma,
                                 float* curF, __bf16* curB, long n) {
  const long i = blockIdx.x * (long)blockDim.x + threadIdx.x;
  if (i >= n) return;
  const float v = gamma[0] * attn[i] + curF[i];
  curF[i] = v;
  curB[i] = (__bf16)v;
}

__global__ void cast_bf16(const float* __restrict__ s, __bf16* __restrict__ d, long n) {
  const long i = blockIdx.x * (long)blockDim.x + threadIdx.x;
  if (i < n) d[i] = (__bf16)s[i];
}

// GroupNorm, one block per (batch, group); in-place on d_out.
__global__ __launch_bounds__(256) void group_norm_k(float* __restrict__ x,
                                                    const float* __restrict__ g,
                                                    const float* __restrict__ b) {
  const int bi = blockIdx.x >> 5;
  const int gi = blockIdx.x & 31;
  float* base = x + (long)bi * C * HWX + (long)gi * (C / NG) * HWX;
  const long n = (long)(C / NG) * HWX;
  float s = 0.0f, ss = 0.0f;
  for (long i = threadIdx.x; i < n; i += 256) {
    const float v = base[i];
    s += v; ss += v * v;
  }
  __shared__ float rs[256], rq[256];
  rs[threadIdx.x] = s; rq[threadIdx.x] = ss;
  __syncthreads();
  for (int o = 128; o; o >>= 1) {
    if (threadIdx.x < o) { rs[threadIdx.x] += rs[threadIdx.x + o];
                           rq[threadIdx.x] += rq[threadIdx.x + o]; }
    __syncthreads();
  }
  const float mu  = rs[0] / (float)n;
  const float var = rq[0] / (float)n - mu * mu;
  const float inv = rsqrtf(var + 1e-5f);
  for (long i = threadIdx.x; i < n; i += 256) {
    const int c = gi * (C / NG) + (int)(i / HWX);
    base[i] = (base[i] - mu) * inv * g[c] + b[c];
  }
}

// ---------------------------------------------------------------------------
static inline void launch_gemm(hipStream_t st, const GemmP& p) {
  dim3 grid((p.N + 127) / 128, (p.M + 127) / 128, p.slices);
  hipLaunchKernelGGL(wmma_gemm, grid, dim3(256), 0, st, p);
}

extern "C" void kernel_launch(void* const* d_in, const int* in_sizes, int n_in,
                              void* d_out, int out_size, void* d_ws, size_t ws_size,
                              hipStream_t stream) {
  const float* x       = (const float*)d_in[0];
  const float* conv1_w = (const float*)d_in[1];
  const float* conv1_b = (const float*)d_in[2];
  const float* q_w     = (const float*)d_in[3];
  const float* q_b     = (const float*)d_in[4];
  const float* k_w     = (const float*)d_in[5];
  const float* k_b     = (const float*)d_in[6];
  const float* v_w     = (const float*)d_in[7];
  const float* v_b     = (const float*)d_in[8];
  const float* gamma   = (const float*)d_in[9];
  const float* cout_w  = (const float*)d_in[10];
  const float* gn_g    = (const float*)d_in[11];
  const float* gn_b    = (const float*)d_in[12];
  float* out = (float*)d_out;

  // Workspace carve-up (256B aligned)
  size_t off = 0;
  auto alloc = [&](size_t bytes) -> void* {
    void* p = (char*)d_ws + off;
    off += (bytes + 255) & ~(size_t)255;
    return p;
  };
  __bf16* xb   = (__bf16*)alloc((size_t)BATCH * CIN * HWX * 2);
  __bf16* w1b  = (__bf16*)alloc((size_t)C * CIN * 2);
  __bf16* qwb  = (__bf16*)alloc((size_t)CQK * C * 2);
  __bf16* kwb  = (__bf16*)alloc((size_t)CQK * C * 2);
  __bf16* vwb  = (__bf16*)alloc((size_t)C * C * 2);
  __bf16* cwb  = (__bf16*)alloc((size_t)C * C * 9 * 2);
  float*  curF = (float*) alloc((size_t)BATCH * C * HWX * 4);
  __bf16* curB = (__bf16*)alloc((size_t)BATCH * C * HWX * 2);
  __bf16* qb   = (__bf16*)alloc((size_t)BATCH * CQK * HWX * 2);
  __bf16* kb   = (__bf16*)alloc((size_t)BATCH * CQK * HWX * 2);
  __bf16* vb   = (__bf16*)alloc((size_t)BATCH * C * HWX * 2);
  float*  E    = (float*) alloc((size_t)BATCH * HWX * 192 * 4);
  __bf16* P    = (__bf16*)alloc((size_t)BATCH * HWX * 192 * 2);
  float*  attn = (float*) alloc((size_t)BATCH * C * HWX * 4);
  (void)ws_size; (void)n_in; (void)in_sizes; (void)out_size;

  auto cast = [&](const float* s, __bf16* d, long n) {
    hipLaunchKernelGGL(cast_bf16, dim3((unsigned)((n + 255) / 256)), dim3(256), 0, stream, s, d, n);
  };
  cast(x,       xb,  (long)BATCH * CIN * HWX);
  cast(conv1_w, w1b, (long)C * CIN);
  cast(q_w,     qwb, (long)CQK * C);
  cast(k_w,     kwb, (long)CQK * C);
  cast(v_w,     vwb, (long)C * C);
  cast(cout_w,  cwb, (long)C * C * 9);

  const long E_SL = HWX * 192;   // per-batch E/P slab

  // --- conv1x1: cur = w1 @ x + b -------------------------------------------
  {
    GemmP p = {};
    p.A = w1b; p.B = xb; p.bias = conv1_b; p.Cf = curF; p.Cb = curB;
    p.Am = CIN; p.Ak = 1; p.Ain = 0; p.Aout = 0;
    p.Bk = HWX; p.Bn = 1; p.Bin = 0; p.Bout = (long)CIN * HWX;
    p.Cm = HWX; p.Cn = 1; p.Cin = 0; p.Cout = (long)C * HWX;
    p.M = C; p.N = (int)HWX; p.K = CIN; p.D0 = 1; p.slices = BATCH;
    p.flags = FLAG_ASYNC_A | FLAG_BN1;
    launch_gemm(stream, p);
  }

  // --- 2x criss-cross attention -------------------------------------------
  for (int it = 0; it < 2; ++it) {
    // q = qw @ cur + qb  (bf16 only)
    GemmP pq = {};
    pq.A = qwb; pq.B = curB; pq.bias = q_b; pq.Cf = nullptr; pq.Cb = qb;
    pq.Am = C; pq.Ak = 1; pq.Ain = 0; pq.Aout = 0;
    pq.Bk = HWX; pq.Bn = 1; pq.Bin = 0; pq.Bout = (long)C * HWX;
    pq.Cm = HWX; pq.Cn = 1; pq.Cin = 0; pq.Cout = (long)CQK * HWX;
    pq.M = CQK; pq.N = (int)HWX; pq.K = C; pq.D0 = 1; pq.slices = BATCH;
    pq.flags = FLAG_ASYNC_A | FLAG_BN1;
    launch_gemm(stream, pq);

    GemmP pk = pq; pk.A = kwb; pk.bias = k_b; pk.Cb = kb;
    launch_gemm(stream, pk);

    GemmP pv = pq; pv.A = vwb; pv.bias = v_b; pv.Cb = vb;
    pv.M = C; pv.Cout = (long)C * HWX;
    launch_gemm(stream, pv);

    // e_h[b,h,w,i] = sum_c q[c,h,w] k[c,i,w]   (slice = (b,w), diag mask)
    GemmP ph = {};
    ph.A = qb; ph.B = kb; ph.Cf = E;
    ph.Am = W; ph.Ak = HWX; ph.Ain = 1;  ph.Aout = (long)CQK * HWX;
    ph.Bk = HWX; ph.Bn = W; ph.Bin = 1;  ph.Bout = (long)CQK * HWX;
    ph.Cm = 192L * W; ph.Cn = 1; ph.Cin = 192; ph.Cout = E_SL;
    ph.M = H; ph.N = H; ph.K = CQK; ph.D0 = W; ph.slices = BATCH * W;
    ph.flags = FLAG_MASK_DIAG;
    launch_gemm(stream, ph);

    // e_w[b,h,w,j] = sum_c q[c,h,w] k[c,h,j]   (slice = (b,h))
    GemmP pw = {};
    pw.A = qb; pw.B = kb; pw.Cf = E + H;   // logits 96..191
    pw.Am = 1; pw.Ak = HWX; pw.Ain = W;  pw.Aout = (long)CQK * HWX;
    pw.Bk = HWX; pw.Bn = 1; pw.Bin = W;  pw.Bout = (long)CQK * HWX;
    pw.Cm = 192; pw.Cn = 1; pw.Cin = 192L * W; pw.Cout = E_SL;
    pw.M = W; pw.N = W; pw.K = CQK; pw.D0 = H; pw.slices = BATCH * H;
    pw.flags = 0;
    launch_gemm(stream, pw);

    // softmax over 192 -> P (bf16)
    {
      const int npix = BATCH * (int)HWX;
      hipLaunchKernelGGL(softmax192, dim3((npix * 32 + 255) / 256), dim3(256), 0,
                         stream, E, P, npix);
    }

    // attn[c,h,w]  = sum_i v[c,i,w] * a_h[h,w,i]   (slice = (b,w))
    GemmP o1 = {};
    o1.A = vb; o1.B = P; o1.Cf = attn;
    o1.Am = HWX; o1.Ak = W; o1.Ain = 1;  o1.Aout = (long)C * HWX;
    o1.Bk = 1; o1.Bn = 192L * W; o1.Bin = 192; o1.Bout = E_SL;
    o1.Cm = HWX; o1.Cn = W; o1.Cin = 1;  o1.Cout = (long)C * HWX;
    o1.M = C; o1.N = H; o1.K = H; o1.D0 = W; o1.slices = BATCH * W; o1.flags = 0;
    launch_gemm(stream, o1);

    // attn[c,h,w] += sum_j v[c,h,j] * a_w[h,w,j]   (slice = (b,h))
    GemmP o2 = {};
    o2.A = vb; o2.B = P + H; o2.Cf = attn;
    o2.Am = HWX; o2.Ak = 1; o2.Ain = W;  o2.Aout = (long)C * HWX;
    o2.Bk = 1; o2.Bn = 192; o2.Bin = 192L * W; o2.Bout = E_SL;
    o2.Cm = HWX; o2.Cn = 1; o2.Cin = W;  o2.Cout = (long)C * HWX;
    o2.M = C; o2.N = W; o2.K = W; o2.D0 = H; o2.slices = BATCH * H;
    o2.flags = FLAG_ACCUM | FLAG_ASYNC_A;
    launch_gemm(stream, o2);

    // cur = gamma*attn + cur (and refresh bf16 mirror)
    {
      const long n = (long)BATCH * C * HWX;
      hipLaunchKernelGGL(combine_residual, dim3((unsigned)((n + 255) / 256)),
                         dim3(256), 0, stream, attn, gamma, curF, curB, n);
    }
  }

  // --- conv3x3 -> d_out (f32) ---------------------------------------------
  {
    dim3 grid((unsigned)(HWX / 128), C / 128, BATCH);
    hipLaunchKernelGGL(wmma_conv3x3, grid, dim3(256), 0, stream, curB, cwb, out);
  }

  // --- group norm, in place on d_out --------------------------------------
  hipLaunchKernelGGL(group_norm_k, dim3(BATCH * NG), dim3(256), 0, stream,
                     out, gn_g, gn_b);
}
